// RNATracker_18897856102690
// MI455X (gfx1250) — matrix operand order repaired and checked by reading, not compiled, not run on hardware
//
#include <hip/hip_runtime.h>
#include <hip/hip_bf16.h>

typedef __attribute__((ext_vector_type(16))) _Float16 v16h;
typedef __attribute__((ext_vector_type(8)))  _Float16 v8h;
typedef __attribute__((ext_vector_type(8)))  float    v8f;

#define NB 2048      // batch
#define NT 128       // LSTM seq length == conv channels
#define NF 11        // LSTM feature size
#define NH 100       // encoder hidden
#define NDH 200      // decoder hidden

__device__ __forceinline__ float sigf(float x) { return 1.f / (1.f + expf(-x)); }

__device__ __forceinline__ v16h cat8(v8h lo, v8h hi) {
    return __builtin_shufflevector(lo, hi, 0, 1, 2, 3, 4, 5, 6, 7,
                                   8, 9, 10, 11, 12, 13, 14, 15);
}

__device__ __forceinline__ void wmma_acc(v8f& c, v16h a, v16h b) {
    c = __builtin_amdgcn_wmma_f32_16x16x32_f16(false, a, false, b, (short)0, c,
                                               false, false);
}

// A-fragment from row-major A: rowBase = &A[row(lane)][0]; two contiguous 8-half
// runs per lane (ISA 7.12.2: lanes 0-15 K{0..7,16..23}, lanes 16-31 K{8..15,24..31}).
__device__ __forceinline__ v16h load_a16(const _Float16* rowBase, int k0) {
    const int half = (threadIdx.x >> 4) & 1;
    const _Float16* p = rowBase + k0 + half * 8;
    return cat8(*(const v8h*)p, *(const v8h*)(p + 16));
}

// B-fragment from pre-packed fragment layout [kb][nb][lane][16]: one 32B load.
__device__ __forceinline__ v16h load_bfrag(const _Float16* Bf, int Ntiles, int nt, int k0) {
    const int lane = threadIdx.x & 31;
    return *(const v16h*)(Bf + ((((size_t)(k0 >> 5)) * Ntiles + nt) << 9) + lane * 16);
}

// ---------------------------------------------------------------------------
// Pack f32 weight W[N][K] into WMMA B-fragment layout (transpose + K-pad):
// dst[((kb*Ntiles+nb)*32+lane)*16+j] = W[nb*16+(lane&15)][kb*32+(j&7)+16*(j>=8)+8*(lane>=16)]
// ---------------------------------------------------------------------------
__global__ void pack_bfrag(_Float16* __restrict__ dst, const float* __restrict__ src,
                           int N, int K, int Kpad) {
    int tid = blockIdx.x * blockDim.x + threadIdx.x;
    if (tid >= N * Kpad) return;
    int j    = tid & 15;
    int lane = (tid >> 4) & 31;
    int blk  = tid >> 9;
    int Ntiles = N >> 4;
    int kb = blk / Ntiles, nb = blk % Ntiles;
    int n = nb * 16 + (lane & 15);
    int k = kb * 32 + (j & 7) + ((j >= 8) ? 16 : 0) + ((lane >= 16) ? 8 : 0);
    dst[tid] = (k < K) ? (_Float16)src[n * K + k] : (_Float16)0.f;
}

// conv2 weights with reordered K: k = kk*128 + ci  (ci contiguous for vector A loads)
__global__ void pack_bfrag_conv2(_Float16* __restrict__ dst, const float* __restrict__ src) {
    int tid = blockIdx.x * blockDim.x + threadIdx.x;
    if (tid >= 128 * 1280) return;
    int j    = tid & 15;
    int lane = (tid >> 4) & 31;
    int blk  = tid >> 9;               // kb*8 + nb (Ntiles = 8)
    int kb = blk >> 3, nb = blk & 7;
    int n = nb * 16 + (lane & 15);     // co
    int k = kb * 32 + (j & 7) + ((j >= 8) ? 16 : 0) + ((lane >= 16) ? 8 : 0);
    int ci = k & 127, kk = k >> 7;
    dst[tid] = (_Float16)src[n * 1280 + ci * 10 + kk];
}

__global__ void add_bias(float* __restrict__ dst, const float* __restrict__ a,
                         const float* __restrict__ b, int n) {
    int tid = blockIdx.x * blockDim.x + threadIdx.x;
    if (tid < n) dst[tid] = a[tid] + b[tid];
}

// ---------------------------------------------------------------------------
// conv1 (4->128, K=10, pad 5) + relu + maxpool3 -> h1pT [B,44,128] f16
// (pool length 34 padded by 5 each side; channel-fastest for conv2 A loads)
// ---------------------------------------------------------------------------
__global__ void conv1_pool(const float* __restrict__ x, const float* __restrict__ w,
                           const float* __restrict__ bias, _Float16* __restrict__ h1pT) {
    int tid = blockIdx.x * blockDim.x + threadIdx.x;
    if (tid >= NB * 44 * NT) return;
    int co = tid & 127;
    int tp = (tid >> 7) % 44;
    int b  = tid / (44 * NT);
    if (tp < 5 || tp >= 39) { h1pT[tid] = (_Float16)0.f; return; }
    int t0 = (tp - 5) * 3;
    float best = 0.f;                  // relu -> max >= 0
#pragma unroll
    for (int j = 0; j < 3; ++j) {
        int t = t0 + j;                // conv output index 0..101
        float acc = bias[co];
        for (int ci = 0; ci < 4; ++ci) {
            const float* xp = x + (b * 4 + ci) * 101;
            const float* wp = w + (co * 4 + ci) * 10;
#pragma unroll
            for (int k = 0; k < 10; ++k) {
                int xi = t + k - 5;
                float xv = (xi >= 0 && xi < 101) ? xp[xi] : 0.f;
                acc += xv * wp[k];
            }
        }
        best = fmaxf(best, fmaxf(acc, 0.f));
    }
    h1pT[tid] = (_Float16)best;
}

// ---------------------------------------------------------------------------
// conv2 WMMA GEMM: M = B*35 rows (b,t), N = 128, K = 1280 with k = kk*128+ci.
// A run of 8 K-values = 8 contiguous channels of h1pT -> v8h loads.
// ---------------------------------------------------------------------------
__global__ void __launch_bounds__(256) conv2_gemm(const _Float16* __restrict__ h1pT,
                                                  const _Float16* __restrict__ Bc2,
                                                  const float* __restrict__ bias2,
                                                  float* __restrict__ y2) {
    const int tile = blockIdx.x * 8 + (threadIdx.x >> 5);
    const int mt = tile >> 3, nt = tile & 7;   // Ntiles = 8
    const int m0 = mt * 16, n0 = nt * 16;
    const int lane = threadIdx.x & 31;
    const int half = lane >> 4, col = lane & 15;
    v8f c;
    float bv = bias2[n0 + col];
#pragma unroll
    for (int r = 0; r < 8; ++r) c[r] = bv;
    const int m = m0 + (lane & 15);
    const int bb = m / 35, tt = m % 35;
    const _Float16* arow = h1pT + ((size_t)bb * 44 + tt) * 128;
    for (int k0 = 0; k0 < 1280; k0 += 32) {
        int s0 = k0 + half * 8, s1 = s0 + 16;
        v8h lo = *(const v8h*)(arow + ((s0 >> 7) << 7) + (s0 & 127));
        v8h hi = *(const v8h*)(arow + ((s1 >> 7) << 7) + (s1 & 127));
        wmma_acc(c, cat8(lo, hi), load_bfrag(Bc2, 8, nt, k0));
    }
#pragma unroll
    for (int r = 0; r < 8; ++r) {
        int mm = m0 + r + (half ? 8 : 0);
        y2[mm * 128 + n0 + col] = fmaxf(c[r], 0.f);
    }
}

// maxpool3 on conv2 output -> h2 [B,128,11] f16 (encoder LSTM input)
__global__ void pool2(const float* __restrict__ y2, _Float16* __restrict__ h2) {
    int tid = blockIdx.x * blockDim.x + threadIdx.x;
    if (tid >= NB * NT * NF) return;
    int b = tid / (NT * NF);
    int rm = tid % (NT * NF);
    int co = rm / NF, u = rm % NF;
    float m = y2[(b * 35 + 3 * u) * 128 + co];
    m = fmaxf(m, y2[(b * 35 + 3 * u + 1) * 128 + co]);
    m = fmaxf(m, y2[(b * 35 + 3 * u + 2) * 128 + co]);
    h2[tid] = (_Float16)m;
}

__global__ void enc_init(_Float16* __restrict__ hbuf, float* __restrict__ cbuf) {
    int tid = blockIdx.x * blockDim.x + threadIdx.x;
    if (tid < 2 * NB * NT) hbuf[tid] = (_Float16)0.f;
    if (tid < 2 * NB * NH) cbuf[tid] = 0.f;
}

// ---------------------------------------------------------------------------
// Encoder per-step gates (dirs via blockIdx.z):
// gates = bias + x_t @ wih^T (K=32, scalar A gather) + h @ whh^T (K=128, vector)
// ---------------------------------------------------------------------------
__global__ void __launch_bounds__(256) enc_gates(const _Float16* __restrict__ h2,
                                                 const _Float16* __restrict__ wihB,
                                                 const _Float16* __restrict__ whhB,
                                                 const float* __restrict__ encbias,
                                                 const _Float16* __restrict__ hbuf,
                                                 float* __restrict__ gates, int s) {
    const int d  = blockIdx.z;
    const int td = d ? (NT - 1 - s) : s;
    const int tile = blockIdx.x * 8 + (threadIdx.x >> 5);
    const int mt = tile / 25, nt = tile % 25;
    const int m0 = mt * 16, n0 = nt * 16;
    const int lane = threadIdx.x & 31;
    const int half = lane >> 4, r = lane & 15, col = lane & 15;
    v8f c;
    float bv = encbias[d * 400 + n0 + col];
#pragma unroll
    for (int rr = 0; rr < 8; ++rr) c[rr] = bv;
    {   // x part: K=32 (11 valid), ragged rows -> scalar gather
        const _Float16* xrow = h2 + ((size_t)(m0 + r) * NT + td) * NF;
        v16h a;
#pragma unroll
        for (int j = 0; j < 16; ++j) {
            int ka = j + ((j >= 8) ? 8 : 0) + (half ? 8 : 0);
            a[j] = (ka < NF) ? xrow[ka] : (_Float16)0.f;
        }
        wmma_acc(c, a, load_bfrag(wihB + d * 12800, 25, nt, 0));
    }
    {   // recurrent part: K=128 (hbuf pad rows are zero)
        const _Float16* arow = hbuf + ((size_t)(d * NB) + m0 + r) * NT;
        const _Float16* Bf = whhB + d * 51200;
        for (int k0 = 0; k0 < 128; k0 += 32)
            wmma_acc(c, load_a16(arow, k0), load_bfrag(Bf, 25, nt, k0));
    }
#pragma unroll
    for (int rr = 0; rr < 8; ++rr) {
        int m = m0 + rr + (half ? 8 : 0);
        gates[(d * NB + m) * 400 + n0 + col] = c[rr];
    }
}

// LSTM cell update + write encoder outputs into xl [B,128,200] f16
__global__ void enc_cell(const float* __restrict__ gates, _Float16* __restrict__ hbuf,
                         float* __restrict__ cbuf, _Float16* __restrict__ xl, int s) {
    int tid = blockIdx.x * blockDim.x + threadIdx.x;
    if (tid >= 2 * NB * NH) return;
    int d = tid / (NB * NH);
    int rm = tid % (NB * NH);
    int b = rm / NH, j = rm % NH;
    const float* g = gates + (size_t)(d * NB + b) * 400;
    float iv = sigf(g[j]);
    float fv = sigf(g[NH + j]);
    float gv = tanhf(g[2 * NH + j]);
    float ov = sigf(g[3 * NH + j]);
    float cold = cbuf[(d * NB + b) * NH + j];
    float cnew = fv * cold + iv * gv;
    float hnew = ov * tanhf(cnew);
    cbuf[(d * NB + b) * NH + j] = cnew;
    hbuf[(d * NB + b) * NT + j] = (_Float16)hnew;
    int td = d ? (NT - 1 - s) : s;
    xl[((size_t)b * NT + td) * NDH + d * NH + j] = (_Float16)hnew;
}

// ---------------------------------------------------------------------------
// Decoder init: faithful PyTorch reshape gather (flat i = d*H*B + j*B + b)
// ---------------------------------------------------------------------------
__global__ void dec_init(const _Float16* __restrict__ hbuf, const float* __restrict__ cbuf,
                         float* __restrict__ hn, _Float16* __restrict__ hnf16,
                         float* __restrict__ cn, _Float16* __restrict__ token) {
    int tid = blockIdx.x * blockDim.x + threadIdx.x;
    if (tid < NB * 416) token[tid] = (_Float16)0.f;
    if (tid < NB * NDH) {
        int i = tid;
        int d = i / (NH * NB);
        int j = (i / NB) % NH;
        int b = i % NB;
        hn[tid] = (float)hbuf[(d * NB + b) * NT + j];
        cn[tid] = cbuf[(d * NB + b) * NH + j];
    }
    if (tid < NB * 224) {
        int r = tid / 224, cc = tid % 224;
        float v = 0.f;
        if (cc < NDH) {
            int i = r * NDH + cc;
            int d = i / (NH * NB);
            int j = (i / NB) % NH;
            int b = i % NB;
            v = (float)hbuf[(d * NB + b) * NT + j];
        }
        hnf16[tid] = (_Float16)v;
    }
}

// ---------------------------------------------------------------------------
// Decoder gates: bias + token[B,416] @ dec_wih^T + hn[B,224] @ dec_whh^T
// ---------------------------------------------------------------------------
__global__ void __launch_bounds__(256) dec_gates(const _Float16* __restrict__ token,
                                                 const _Float16* __restrict__ wihB,
                                                 const _Float16* __restrict__ hnf16,
                                                 const _Float16* __restrict__ whhB,
                                                 const float* __restrict__ bias,
                                                 float* __restrict__ gates) {
    const int tile = blockIdx.x * 8 + (threadIdx.x >> 5);
    const int mt = tile / 50, nt = tile % 50;
    const int m0 = mt * 16, n0 = nt * 16;
    const int lane = threadIdx.x & 31;
    const int half = lane >> 4, r = lane & 15, col = lane & 15;
    v8f c;
    float bv = bias[n0 + col];
#pragma unroll
    for (int rr = 0; rr < 8; ++rr) c[rr] = bv;
    {
        const _Float16* arow = token + (size_t)(m0 + r) * 416;
        for (int k0 = 0; k0 < 416; k0 += 32)
            wmma_acc(c, load_a16(arow, k0), load_bfrag(wihB, 50, nt, k0));
    }
    {
        const _Float16* arow = hnf16 + (size_t)(m0 + r) * 224;
        for (int k0 = 0; k0 < 224; k0 += 32)
            wmma_acc(c, load_a16(arow, k0), load_bfrag(whhB, 50, nt, k0));
    }
#pragma unroll
    for (int rr = 0; rr < 8; ++rr) {
        int m = m0 + rr + (half ? 8 : 0);
        gates[(size_t)m * 800 + n0 + col] = c[rr];
    }
}

__global__ void dec_cell(const float* __restrict__ gates, float* __restrict__ hn,
                         _Float16* __restrict__ hnf16, float* __restrict__ cn) {
    int tid = blockIdx.x * blockDim.x + threadIdx.x;
    if (tid >= NB * NDH) return;
    int b = tid / NDH, j = tid % NDH;
    const float* g = gates + (size_t)b * 800;
    float iv = sigf(g[j]);
    float fv = sigf(g[NDH + j]);
    float gv = tanhf(g[2 * NDH + j]);
    float ov = sigf(g[3 * NDH + j]);
    float cnew = fv * cn[tid] + iv * gv;
    float hnew = ov * tanhf(cnew);
    cn[tid] = cnew;
    hn[tid] = hnew;
    hnf16[b * 224 + j] = (_Float16)hnew;
}

// Attention over xl (f16, L2-resident): scores -> softmax -> context; new token.
__global__ void __launch_bounds__(128) attn(const _Float16* __restrict__ xl,
                                            const float* __restrict__ hn,
                                            _Float16* __restrict__ token) {
    __shared__ float sc[NT];
    __shared__ float red[NT];
    const int b = blockIdx.x;
    const int t = threadIdx.x;
    const _Float16* xb = xl + (size_t)b * NT * NDH;
    float acc = 0.f;
    const v8h* xr = (const v8h*)(xb + t * NDH);
    for (int q = 0; q < NDH / 8; ++q) {
        v8h xv = xr[q];
#pragma unroll
        for (int u = 0; u < 8; ++u) acc += (float)xv[u] * hn[b * NDH + q * 8 + u];
    }
    sc[t] = acc; red[t] = acc;
    __syncthreads();
    for (int s = 64; s > 0; s >>= 1) { if (t < s) red[t] = fmaxf(red[t], red[t + s]); __syncthreads(); }
    float mx = red[0];
    __syncthreads();
    float e = expf(sc[t] - mx);
    sc[t] = e; red[t] = e;
    __syncthreads();
    for (int s = 64; s > 0; s >>= 1) { if (t < s) red[t] += red[t + s]; __syncthreads(); }
    float inv = 1.f / red[0];
    __syncthreads();
    sc[t] *= inv;
    __syncthreads();
    for (int dd = t; dd < NDH; dd += 128) {
        float cacc = 0.f;
        for (int tt = 0; tt < NT; ++tt) cacc += sc[tt] * (float)xb[tt * NDH + dd];
        token[(size_t)b * 416 + dd]       = (_Float16)cacc;
        token[(size_t)b * 416 + NDH + dd] = (_Float16)hn[b * NDH + dd];
    }
}

__global__ void final_out(const _Float16* __restrict__ token, const float* __restrict__ ow,
                          const float* __restrict__ ob, float* __restrict__ out) {
    int b = blockIdx.x * blockDim.x + threadIdx.x;
    if (b >= NB) return;
    float acc = ob[0];
    const v8h* tr = (const v8h*)(token + (size_t)b * 416);
    for (int q = 0; q < 50; ++q) {
        v8h tv = tr[q];
#pragma unroll
        for (int u = 0; u < 8; ++u) acc += (float)tv[u] * ow[q * 8 + u];
    }
    out[b] = sigf(acc);
}

// ---------------------------------------------------------------------------
extern "C" void kernel_launch(void* const* d_in, const int* in_sizes, int n_in,
                              void* d_out, int out_size, void* d_ws, size_t ws_size,
                              hipStream_t stream) {
    (void)in_sizes; (void)n_in; (void)out_size; (void)ws_size;
    const float* x       = (const float*)d_in[0];
    const float* c1w     = (const float*)d_in[1];
    const float* c1b     = (const float*)d_in[2];
    const float* c2w     = (const float*)d_in[3];
    const float* c2b     = (const float*)d_in[4];
    const float* wih_f   = (const float*)d_in[5];
    const float* whh_f   = (const float*)d_in[6];
    const float* bih_f   = (const float*)d_in[7];
    const float* bhh_f   = (const float*)d_in[8];
    const float* wih_b   = (const float*)d_in[9];
    const float* whh_b   = (const float*)d_in[10];
    const float* bih_b   = (const float*)d_in[11];
    const float* bhh_b   = (const float*)d_in[12];
    const float* dwih    = (const float*)d_in[13];
    const float* dwhh    = (const float*)d_in[14];
    const float* dbih    = (const float*)d_in[15];
    const float* dbhh    = (const float*)d_in[16];
    const float* ow      = (const float*)d_in[17];
    const float* ob      = (const float*)d_in[18];
    float* out = (float*)d_out;

    char* base = (char*)d_ws;
    size_t off = 0;
    auto alloc = [&](size_t bytes) {
        size_t o = off;
        off = (off + bytes + 255) & ~(size_t)255;
        return o;
    };
    _Float16* h1pT    = (_Float16*)(base + alloc((size_t)NB * 44 * NT * 2));
    _Float16* Bc2     = (_Float16*)(base + alloc((size_t)1280 * 128 * 2));
    _Float16* h2      = (_Float16*)(base + alloc((size_t)NB * NT * NF * 2));
    _Float16* wihB    = (_Float16*)(base + alloc((size_t)2 * 32 * 400 * 2));
    _Float16* whhB    = (_Float16*)(base + alloc((size_t)2 * 128 * 400 * 2));
    float*    encbias = (float*)   (base + alloc((size_t)2 * 400 * 4));
    _Float16* hbuf    = (_Float16*)(base + alloc((size_t)2 * NB * NT * 2));
    float*    cbuf    = (float*)   (base + alloc((size_t)2 * NB * NH * 4));
    float*    gatesE  = (float*)   (base + alloc((size_t)2 * NB * 400 * 4));
    // big region: y2 [B*35,128] f32 (36.7 MB) then reused as xl [B,128,200] f16 (105 MB)
    size_t bigOff = alloc((size_t)NB * NT * NDH * 2);
    float*    y2 = (float*)(base + bigOff);
    _Float16* xl = (_Float16*)(base + bigOff);
    _Float16* dwihB   = (_Float16*)(base + alloc((size_t)416 * 800 * 2));
    _Float16* dwhhB   = (_Float16*)(base + alloc((size_t)224 * 800 * 2));
    float*    decbias = (float*)   (base + alloc((size_t)800 * 4));
    _Float16* token   = (_Float16*)(base + alloc((size_t)NB * 416 * 2));
    float*    hn      = (float*)   (base + alloc((size_t)NB * NDH * 4));
    _Float16* hnf16   = (_Float16*)(base + alloc((size_t)NB * 224 * 2));
    float*    cn      = (float*)   (base + alloc((size_t)NB * NDH * 4));
    float*    gatesD  = (float*)   (base + alloc((size_t)NB * 800 * 4));

    auto g1 = [](int n) { return dim3((n + 255) / 256); };

    // ---- weight packing into WMMA B-fragment layouts ----
    pack_bfrag_conv2<<<g1(128 * 1280), 256, 0, stream>>>(Bc2, c2w);
    pack_bfrag<<<g1(400 * 32), 256, 0, stream>>>(wihB,          wih_f, 400, NF, 32);
    pack_bfrag<<<g1(400 * 32), 256, 0, stream>>>(wihB + 12800,  wih_b, 400, NF, 32);
    pack_bfrag<<<g1(400 * 128), 256, 0, stream>>>(whhB,          whh_f, 400, NH, 128);
    pack_bfrag<<<g1(400 * 128), 256, 0, stream>>>(whhB + 51200,  whh_b, 400, NH, 128);
    pack_bfrag<<<g1(800 * 416), 256, 0, stream>>>(dwihB, dwih, 800, 400, 416);
    pack_bfrag<<<g1(800 * 224), 256, 0, stream>>>(dwhhB, dwhh, 800, 200, 224);
    add_bias<<<g1(400), 256, 0, stream>>>(encbias,       bih_f, bhh_f, 400);
    add_bias<<<g1(400), 256, 0, stream>>>(encbias + 400, bih_b, bhh_b, 400);
    add_bias<<<g1(800), 256, 0, stream>>>(decbias, dbih, dbhh, 800);

    // ---- conv stack ----
    conv1_pool<<<g1(NB * 44 * NT), 256, 0, stream>>>(x, c1w, c1b, h1pT);
    conv2_gemm<<<dim3(4480), 256, 0, stream>>>(h1pT, Bc2, c2b, y2);   // 35840 tiles
    pool2<<<g1(NB * NT * NF), 256, 0, stream>>>(y2, h2);

    // ---- bidirectional encoder LSTM (serial over 128 steps) ----
    enc_init<<<g1(2 * NB * NT), 256, 0, stream>>>(hbuf, cbuf);
    for (int s = 0; s < NT; ++s) {
        enc_gates<<<dim3(400, 1, 2), 256, 0, stream>>>(h2, wihB, whhB, encbias,
                                                       hbuf, gatesE, s);
        enc_cell<<<g1(2 * NB * NH), 256, 0, stream>>>(gatesE, hbuf, cbuf, xl, s);
    }

    // ---- decoder with attention (10 serial steps) ----
    dec_init<<<g1(NB * 416), 256, 0, stream>>>(hbuf, cbuf, hn, hnf16, cn, token);
    for (int s = 0; s < 10; ++s) {
        dec_gates<<<dim3(800), 256, 0, stream>>>(token, dwihB, hnf16, dwhhB,
                                                 decbias, gatesD);      // 6400 tiles
        dec_cell<<<g1(NB * NDH), 256, 0, stream>>>(gatesD, hn, hnf16, cn);
        attn<<<dim3(NB), 128, 0, stream>>>(xl, hn, token);
    }
    final_out<<<g1(NB), 256, 0, stream>>>(token, ow, ob, out);
}
